// VAN_CNN_58703613001938
// MI455X (gfx1250) — matrix operand order repaired and checked by reading, not compiled, not run on hardware
//
#include <hip/hip_runtime.h>
#include <hip/hip_bf16.h>
#include <math.h>

// VAN autoregressive sampler for MI455X (gfx1250, wave32).
//  - a1[pix][co][b] (f32, 128MB, L2-resident) maintained incrementally.
//  - conv2 computed only at the 25 needed pixels per step as implicit GEMM
//    (M=batch, N=32 co, K=800) with v_wmma_f32_16x16x32_f16.
//  - Round-2 fix: epilogue fully unrolled (no runtime indexing of the
//    accumulator array -> no scratch spill) and single-barrier-pair
//    LDS transpose epilogue.

#define B_      4096
#define LSIDE   16
#define NPIX    256
#define HCH     32
#define TAPS    25
#define NSTEPS  64
#define EPS_    1e-7f
#define LRELU_  0.1f

#define BT      64      // batch tile per conv2 block
#define NW      8       // waves per conv2 block; waves 0-3 -> co 0..15,
                        // waves 4-7 -> co 16..31, (w&3)*16 batch offset
#define RS      40      // padded LDS row stride in halfs (80B: conflict-free)

typedef __attribute__((ext_vector_type(16))) _Float16 v16h;
typedef __attribute__((ext_vector_type(8)))  _Float16 v8h;
typedef __attribute__((ext_vector_type(8)))  float    v8f;

__device__ __forceinline__ unsigned rotl32(unsigned v, int r) {
  return (v << r) | (v >> (32 - r));
}

// threefry2x32, 20 rounds; returns x0.
__device__ __forceinline__ unsigned threefry2x32(unsigned k0, unsigned k1,
                                                 unsigned c0, unsigned c1) {
  unsigned ks2 = 0x1BD11BDAu ^ k0 ^ k1;
  unsigned x0 = c0 + k0, x1 = c1 + k1;
  x0 += x1; x1 = rotl32(x1, 13) ^ x0;  x0 += x1; x1 = rotl32(x1, 15) ^ x0;
  x0 += x1; x1 = rotl32(x1, 26) ^ x0;  x0 += x1; x1 = rotl32(x1,  6) ^ x0;
  x0 += k1;  x1 += ks2 + 1u;
  x0 += x1; x1 = rotl32(x1, 17) ^ x0;  x0 += x1; x1 = rotl32(x1, 29) ^ x0;
  x0 += x1; x1 = rotl32(x1, 16) ^ x0;  x0 += x1; x1 = rotl32(x1, 24) ^ x0;
  x0 += ks2; x1 += k0 + 2u;
  x0 += x1; x1 = rotl32(x1, 13) ^ x0;  x0 += x1; x1 = rotl32(x1, 15) ^ x0;
  x0 += x1; x1 = rotl32(x1, 26) ^ x0;  x0 += x1; x1 = rotl32(x1,  6) ^ x0;
  x0 += k0;  x1 += k1 + 3u;
  x0 += x1; x1 = rotl32(x1, 17) ^ x0;  x0 += x1; x1 = rotl32(x1, 29) ^ x0;
  x0 += x1; x1 = rotl32(x1, 16) ^ x0;  x0 += x1; x1 = rotl32(x1, 24) ^ x0;
  x0 += k1;  x1 += ks2 + 4u;
  x0 += x1; x1 = rotl32(x1, 13) ^ x0;  x0 += x1; x1 = rotl32(x1, 15) ^ x0;
  x0 += x1; x1 = rotl32(x1, 26) ^ x0;  x0 += x1; x1 = rotl32(x1,  6) ^ x0;
  x0 += ks2; x1 += k0 + 5u;
  (void)x1;
  return x0;
}

// ---------------------------------------------------------------------------
// Kernel 1: one-shot prep. Copies x -> output & transposed working copy,
// zeros logq, repacks weights: w2h[tap][co][ci] (f16), w1s/w3s[tap][ch] (f32).
// ---------------------------------------------------------------------------
__global__ void van_prep_kernel(const float* __restrict__ xin,
                                const float* __restrict__ w1,
                                const float* __restrict__ w2,
                                const float* __restrict__ w3,
                                float* __restrict__ xout,
                                float* __restrict__ logq,
                                float* __restrict__ xT,
                                _Float16* __restrict__ w2h,
                                float* __restrict__ w1s,
                                float* __restrict__ w3s) {
  long long idx = (long long)blockIdx.x * blockDim.x + threadIdx.x;
  const long long N1 = (long long)B_ * NPIX;
  if (idx < N1) {
    int b = (int)(idx >> 8);        // NPIX == 256
    int pix = (int)(idx & 255);
    float v = xin[idx];
    xout[idx] = v;
    xT[(long long)pix * B_ + b] = v;
    return;
  }
  idx -= N1;
  if (idx < B_) { logq[idx] = 0.0f; return; }
  idx -= B_;
  if (idx < (long long)TAPS * HCH * HCH) {
    int t  = (int)(idx / (HCH * HCH));
    int r  = (int)(idx % (HCH * HCH));
    int co = r / HCH, ci = r % HCH;
    w2h[(t * HCH + co) * HCH + ci] =
        (_Float16)w2[((long long)co * HCH + ci) * TAPS + t];
    return;
  }
  idx -= (long long)TAPS * HCH * HCH;
  if (idx < TAPS * HCH) {
    int t = (int)(idx / HCH), co = (int)(idx % HCH);
    w1s[t * HCH + co] = w1[co * TAPS + t];
    return;
  }
  idx -= TAPS * HCH;
  if (idx < TAPS * HCH) {
    int t = (int)(idx / HCH), ci = (int)(idx % HCH);
    w3s[t * HCH + ci] = w3[ci * TAPS + t];
  }
}

// ---------------------------------------------------------------------------
// Kernel 2: full conv1 pre-activation init: a1[pix][co][b] (f32).
// ---------------------------------------------------------------------------
__global__ void van_conv1_init_kernel(const float* __restrict__ xT,
                                      const float* __restrict__ w1s,
                                      const float* __restrict__ b1,
                                      float* __restrict__ a1) {
  __shared__ float xs[TAPS][256];
  const int pix = blockIdx.x;
  const int b0  = blockIdx.y * 256;
  const int tid = threadIdx.x;
  const int pi = pix >> 4, pj = pix & 15;
  for (int u = tid; u < TAPS * 256; u += 256) {
    int t = u >> 8, bl = u & 255;
    int ki = t / 5, kj = t % 5;
    int ip = (((pi + ki - 2) & 15) << 4) | ((pj + kj - 2) & 15);
    xs[t][bl] = xT[(long long)ip * B_ + b0 + bl];
  }
  __syncthreads();
  #pragma unroll 4
  for (int co = 0; co < HCH; ++co) {
    float acc = b1[co];
    #pragma unroll
    for (int t = 0; t < TAPS; ++t) acc += w1s[t * HCH + co] * xs[t][tid];
    a1[(long long)(pix * HCH + co) * B_ + b0 + tid] = acc;
  }
}

// ---------------------------------------------------------------------------
// Kernel 3: conv2 at the 25 pixels around (ci_, cj_) via WMMA implicit GEMM.
// Grid: (5 output rows, B/BT batch groups), 8 waves.
// Wave w: co-tile nt = w>>2, batches wb = (w&3)*16; acc = 5 x v8f (40 VGPRs).
// ---------------------------------------------------------------------------
__global__ void __launch_bounds__(NW * 32, 1)
van_conv2_kernel(const float* __restrict__ a1,
                 const _Float16* __restrict__ w2h,
                 const float* __restrict__ b2,
                 _Float16* __restrict__ h2,
                 int ci_, int cj_) {
  // [col 0..8][bl 0..BT-1] rows of RS halfs (ci in first 32), lrelu'd f16.
  __shared__ __align__(32) _Float16 inbuf[9 * BT * RS];   // 46,080 B

  const int tid  = threadIdx.x;
  const int wave = tid >> 5;
  const int lane = tid & 31;
  const int n    = lane & 15;
  const int hi   = lane >> 4;
  const int nt   = wave >> 2;             // co tile: 0 or 1
  const int wb   = (wave & 3) * 16;       // batch offset within tile
  const int orow = blockIdx.x;            // output row 0..4
  const int b0   = blockIdx.y * BT;
  const int pi   = (ci_ + orow - 2) & 15; // output pixel row

  v8f acc[5];
  #pragma unroll
  for (int oc = 0; oc < 5; ++oc) { v8f z = {}; acc[oc] = z; }

  #pragma unroll 1
  for (int ki = 0; ki < 5; ++ki) {
    const int irow = (pi + ki - 2) & 15;
    __syncthreads();
    // Stage one input row, 9 columns, f16 with leaky-relu applied.
    for (int u = tid; u < 9 * HCH * BT; u += NW * 32) {
      int c  = u >> 11;           // / (HCH*BT) == /2048
      int r  = u & 2047;
      int ch = r >> 6;            // / BT
      int bl = r & (BT - 1);
      int jc = (cj_ + c - 4) & 15;
      float v = a1[(long long)(((irow << 4) | jc) * HCH + ch) * B_ + b0 + bl];
      v = v > 0.0f ? v : LRELU_ * v;
      inbuf[(c * BT + bl) * RS + ch] = (_Float16)v;
    }
    __syncthreads();

    #pragma unroll 1
    for (int kj = 0; kj < 5; ++kj) {
      const int t = ki * 5 + kj;
      // B fragment: 32(K=ci) x 16(N=co); lane holds N=n, K = hi*16 + e.
      const v16h bf =
          *(const v16h*)(w2h + ((t * HCH + nt * 16 + n) * HCH + hi * 16));
      #pragma unroll
      for (int oc = 0; oc < 5; ++oc) {
        const int c = oc + kj;    // staged input column index 0..8
        // A fragment: 16(M=batch) x 32(K=ci); lane holds M=n,
        // K = hi*16 + e (ci = hi*16 .. +15).
        const _Float16* ap = inbuf + (c * BT + wb + n) * RS;
        v8h alo = *(const v8h*)(ap + hi * 16);
        v8h ahi = *(const v8h*)(ap + hi * 16 + 8);
        v16h af = __builtin_shufflevector(alo, ahi, 0, 1, 2, 3, 4, 5, 6, 7,
                                          8, 9, 10, 11, 12, 13, 14, 15);
        acc[oc] = __builtin_amdgcn_wmma_f32_16x16x32_f16(
            false, af, false, bf, (short)0, acc[oc], false, false);
      }
    }
  }

  // Bias + LDS transpose (disjoint region per oc) + coalesced f16 store.
  // tbuf: [oc 0..4][bl 0..BT-1][co 0..31] halfs = 20KB, reuses inbuf.
  const float bias = b2[nt * 16 + n];
  _Float16* tbuf = inbuf;
  __syncthreads();
  #pragma unroll
  for (int oc = 0; oc < 5; ++oc) {
    #pragma unroll
    for (int e = 0; e < 8; ++e) {
      // C/D layout: element e -> M = e + 8*hi; lane n -> N.
      int bl = wb + e + hi * 8;
      tbuf[(oc * BT + bl) * HCH + nt * 16 + n] = (_Float16)(acc[oc][e] + bias);
    }
  }
  __syncthreads();
  // h2[opix][co][b], b contiguous across lanes.
  for (int u = tid; u < 5 * HCH * BT; u += NW * 32) {
    int oc = u >> 11;             // / (HCH*BT)
    int r  = u & 2047;
    int co = r >> 6;              // / BT
    int bl = r & (BT - 1);
    const int opix = orow * 5 + oc;
    h2[(long long)(opix * HCH + co) * B_ + b0 + bl] =
        tbuf[(oc * BT + bl) * HCH + co];
  }
}

// ---------------------------------------------------------------------------
// Kernel 4: conv3 at (ci_,cj_) + sigmoid + sample + logq + incremental a1.
// ---------------------------------------------------------------------------
__global__ void van_conv3_sample_kernel(const _Float16* __restrict__ h2,
                                        const float* __restrict__ w3s,
                                        const float* __restrict__ b3,
                                        const float* __restrict__ w1s,
                                        float* __restrict__ a1,
                                        float* __restrict__ xout,
                                        float* __restrict__ logq,
                                        int ci_, int cj_, int step) {
  const int b = blockIdx.x * blockDim.x + threadIdx.x;
  float acc = b3[0];
  for (int op = 0; op < TAPS; ++op) {
    #pragma unroll
    for (int ch = 0; ch < HCH; ++ch)
      acc += w3s[op * HCH + ch] *
             (float)h2[(long long)(op * HCH + ch) * B_ + b];
  }
  const float p = 1.0f / (1.0f + __expf(-acc));
  const unsigned r = threefry2x32(42u, (unsigned)step, (unsigned)b, 0u);
  const float u = (float)(r >> 8) * (1.0f / 16777216.0f);
  const float samp = (u < p) ? 1.0f : -1.0f;

  const int pix = (ci_ << 4) | cj_;
  const long long xi = (long long)b * NPIX + pix;
  const float old = xout[xi];
  xout[xi] = samp;
  const float delta = samp - old;

  const float xb = (samp + 1.0f) * 0.5f;
  logq[b] += __logf(p + EPS_) * xb + __logf(1.0f - p + EPS_) * (1.0f - xb);

  // Incremental conv1 update: a1 += delta * w1 over the 25 affected pixels.
  for (int t = 0; t < TAPS; ++t) {
    const int ki = t / 5, kj = t % 5;
    const int pp = (((ci_ - ki + 2) & 15) << 4) | ((cj_ - kj + 2) & 15);
    #pragma unroll
    for (int co = 0; co < HCH; ++co)
      a1[(long long)(pp * HCH + co) * B_ + b] += delta * w1s[t * HCH + co];
  }
}

// ---------------------------------------------------------------------------
extern "C" void kernel_launch(void* const* d_in, const int* in_sizes, int n_in,
                              void* d_out, int out_size, void* d_ws,
                              size_t ws_size, hipStream_t stream) {
  (void)in_sizes; (void)n_in; (void)out_size; (void)ws_size;
  const float* x  = (const float*)d_in[0];
  const float* w1 = (const float*)d_in[1];
  const float* b1 = (const float*)d_in[2];
  const float* w2 = (const float*)d_in[3];
  const float* b2 = (const float*)d_in[4];
  const float* w3 = (const float*)d_in[5];
  const float* b3 = (const float*)d_in[6];

  float* xout = (float*)d_out;                 // (B,1,16,16) flat
  float* logq = xout + (size_t)B_ * NPIX;      // (B,)

  // Workspace carve-out (~139 MB): a1 L2-resident is the whole point.
  char* ws = (char*)d_ws;
  size_t off = 0;
  auto carve = [&](size_t bytes) {
    char* p = ws + off;
    off = (off + bytes + 255) & ~(size_t)255;
    return p;
  };
  float*    a1  = (float*)   carve((size_t)NPIX * HCH * B_ * 4); // 128 MB
  float*    xT  = (float*)   carve((size_t)NPIX * B_ * 4);       // 4 MB
  _Float16* h2  = (_Float16*)carve((size_t)TAPS * HCH * B_ * 2); // 6.5 MB
  _Float16* w2h = (_Float16*)carve((size_t)TAPS * HCH * HCH * 2);
  float*    w1s = (float*)   carve((size_t)TAPS * HCH * 4);
  float*    w3s = (float*)   carve((size_t)TAPS * HCH * 4);

  const long long total = (long long)B_ * NPIX + B_ +
                          (long long)TAPS * HCH * HCH + 2 * TAPS * HCH;
  van_prep_kernel<<<(int)((total + 255) / 256), 256, 0, stream>>>(
      x, w1, w2, w3, xout, logq, xT, w2h, w1s, w3s);
  van_conv1_init_kernel<<<dim3(NPIX, B_ / 256), 256, 0, stream>>>(
      xT, w1s, b1, a1);

  for (int s = 0; s < NSTEPS; ++s) {
    const int pi = 2 * (s >> 3) + 1;   // odd-odd sites, row-major (LEVEL 0)
    const int pj = 2 * (s & 7) + 1;
    van_conv2_kernel<<<dim3(5, B_ / BT), NW * 32, 0, stream>>>(
        a1, w2h, b2, h2, pi, pj);
    van_conv3_sample_kernel<<<dim3(B_ / 256), 256, 0, stream>>>(
        h2, w3s, b3, w1s, a1, xout, logq, pi, pj, s);
  }
}
